// InteractionBlock_84851373899787
// MI455X (gfx1250) — compile-verified, matrix operand order checked by reading, not run on hardware
//
#include <hip/hip_runtime.h>

typedef _Float16 half_t;
typedef __attribute__((ext_vector_type(16))) _Float16 v16h;
typedef __attribute__((ext_vector_type(8)))  _Float16 v8h;
typedef __attribute__((ext_vector_type(2)))  _Float16 v2h;
typedef __attribute__((ext_vector_type(8)))  float    v8f;

#define B_ 16
#define N_ 192
#define H_ 128
#define F_ 128
#define G_ 50
#define KP 64   // G padded to WMMA K granularity

// ---- optional gfx1250 async global->LDS staging (compile-safe fallback) ----
#if defined(__AMDGCN__) && defined(__gfx1250__)
#if __has_builtin(__builtin_amdgcn_global_load_async_to_lds_b128) && \
    __has_builtin(__builtin_amdgcn_s_wait_asynccnt)
#define HAS_ASYNC_LDS 1
// builtin signature (from clang diagnostic): (v4i AS(1)*, v4i AS(3)*, Imm, Imm)
typedef int v4i __attribute__((vector_size(4 * sizeof(int))));
typedef __attribute__((address_space(1))) v4i ga_v4i;
typedef __attribute__((address_space(3))) v4i la_v4i;
#endif
#endif

// ---------------- shared memory layout for the fused kernel ----------------
struct SMem {
  alignas(32) half_t rbfh[N_ * KP];   // rbf[b,i,:,:] as f16, K-padded   24.0 KB
  alignas(32) half_t acth[N_ * F_];   // softplus(rbf@W1+b1) as f16      48.0 KB
  alignas(32) half_t w1t[F_ * KP];    // W_rbf1^T (f x g), padded        16.0 KB
  alignas(32) half_t w2t[F_ * F_];    // W_rbf2^T (f x k)                32.0 KB
  alignas(32) float maskf[N_];
  alignas(32) float brbf1[F_];
  alignas(32) float brbf2[F_];
  alignas(32) float agg[F_];
  alignas(32) float h1[H_];
};

// softplus via raw V_EXP_F32 / V_LOG_F32 (base-2); log arg is in (1,2] so the
// hardware ops are accurate and need no denorm fixups.
__device__ __forceinline__ float softplus_f(float x) {
  const float LOG2E = 1.4426950408889634f;
  const float LN2   = 0.6931471805599453f;
  float e = __builtin_amdgcn_exp2f(-fabsf(x) * LOG2E);   // e^{-|x|} in (0,1]
  return fmaxf(x, 0.f) + LN2 * __builtin_amdgcn_logf(1.f + e);
}

// Value-splitting 16-lane reduction: sum p[0..7] over the 16 lanes of this
// lane's half (bit4 fixed, xor masks <= 8 stay inside the half). Each round
// halves the live values per lane; lane ends owning channel
// v = lane[3]*4 + lane[2]*2 + lane[1]; even lanes commit one ds_add_f32.
__device__ __forceinline__ void reduce16_accum(float p[8], int lane,
                                               float* aggbase) {
  const bool b3 = (lane & 8) != 0;
  float q4[4];
  #pragma unroll
  for (int t = 0; t < 4; ++t) {
    float keep = b3 ? p[4 + t] : p[t];
    float send = b3 ? p[t]     : p[4 + t];
    q4[t] = keep + __shfl_xor(send, 8, 32);
  }
  const bool b2 = (lane & 4) != 0;
  float q2[2];
  #pragma unroll
  for (int t = 0; t < 2; ++t) {
    float keep = b2 ? q4[2 + t] : q4[t];
    float send = b2 ? q4[t]     : q4[2 + t];
    q2[t] = keep + __shfl_xor(send, 4, 32);
  }
  const bool b1 = (lane & 2) != 0;
  float keep = b1 ? q2[1] : q2[0];
  float send = b1 ? q2[0] : q2[1];
  float q1 = keep + __shfl_xor(send, 2, 32);
  q1 += __shfl_xor(q1, 1, 32);
  if ((lane & 1) == 0) {
    const int v = ((lane & 8) >> 1) | ((lane & 4) >> 1) | ((lane & 2) >> 1);
    atomicAdd(aggbase + v, q1);
  }
}

// --------- prep: transpose + convert weights into workspace ---------
__global__ void prep_weights(const float* __restrict__ W1,   // [G,F]
                             const float* __restrict__ W2,   // [F,F]
                             const float* __restrict__ Wo1,  // [F,H]
                             const float* __restrict__ Wo2,  // [H,H]
                             half_t* __restrict__ w1t,       // [F,KP] f16
                             half_t* __restrict__ w2t,       // [F,F]  f16
                             float*  __restrict__ wo1t,      // [H,F]  f32 (row = out ch)
                             float*  __restrict__ wo2t) {    // [H,H]  f32
  for (int idx = threadIdx.x; idx < F_ * KP; idx += blockDim.x) {
    int f = idx >> 6, g = idx & 63;
    w1t[idx] = (half_t)(g < G_ ? W1[g * F_ + f] : 0.0f);
  }
  for (int idx = threadIdx.x; idx < F_ * F_; idx += blockDim.x) {
    int f = idx >> 7, k = idx & 127;
    w2t[idx] = (half_t)W2[k * F_ + f];
  }
  for (int idx = threadIdx.x; idx < H_ * F_; idx += blockDim.x) {
    int t = idx >> 7, k = idx & 127;
    wo1t[idx] = Wo1[k * H_ + t];
    wo2t[idx] = Wo2[k * H_ + t];
  }
}

// --------- prep: nf[b,j,:] = features @ W_f + b_f, stored f16 in ws ---------
__global__ void prep_nf(const float* __restrict__ features, // [B*N, H]
                        const float* __restrict__ W_f,      // [H, F]
                        const float* __restrict__ b_f,      // [F]
                        half_t* __restrict__ nfh) {         // [B*N, F]
  const int r = blockIdx.x;     // b*N + j
  const int t = threadIdx.x;    // 0..127
  __shared__ float frow[H_];
  frow[t] = features[(size_t)r * H_ + t];
  __syncthreads();
  float acc = b_f[t];
  #pragma unroll 4
  for (int k = 0; k < H_; ++k) acc = fmaf(frow[k], W_f[k * F_ + t], acc);
  nfh[(size_t)r * F_ + t] = (half_t)acc;
}

// ------------- fused per-(b,i) interaction block, WMMA f16 path -------------
// 12 waves: wave = (j-pair jp in 0..5) x (f-half fh in 0..1).
// Each wave computes 2 j-strips x 4 f-tiles so every weight A-fragment feeds
// two WMMAs (halves the dominant w2t/w1t LDS fragment traffic).
__global__ __launch_bounds__(384)
void interaction_main(const float* __restrict__ features,
                      const float* __restrict__ rbf,
                      const int*   __restrict__ nmask,
                      const float* __restrict__ b_rbf1,
                      const float* __restrict__ b_rbf2,
                      const float* __restrict__ b_o1,
                      const float* __restrict__ b_o2,
                      const half_t* __restrict__ nfh,
                      const half_t* __restrict__ w1t_g,
                      const half_t* __restrict__ w2t_g,
                      const float* __restrict__ wo1t,
                      const float* __restrict__ wo2t,
                      float* __restrict__ out) {
  __shared__ SMem sm;
  const int i   = blockIdx.x;
  const int b   = blockIdx.y;
  const int tid = threadIdx.x;          // 0..383 (12 wave32s)

  // warm the L2/L0 path for the tail matvecs (global_prefetch_b8)
  if (tid < H_) {
    __builtin_prefetch(&wo1t[tid * F_], 0, 1);
    __builtin_prefetch(&wo2t[tid * F_], 0, 1);
  }

  // ---------------- stage into LDS ----------------
  // filter weights: raw 16B copies; async global->LDS path when available
  {
    const uint4* s1 = (const uint4*)w1t_g;  uint4* d1 = (uint4*)sm.w1t;
    const uint4* s2 = (const uint4*)w2t_g;  uint4* d2 = (uint4*)sm.w2t;
#if defined(HAS_ASYNC_LDS)
    for (int idx = tid; idx < (F_ * KP) / 8; idx += 384)
      __builtin_amdgcn_global_load_async_to_lds_b128(
          (ga_v4i*)(s1 + idx), (la_v4i*)(d1 + idx), 0, 0);
    for (int idx = tid; idx < (F_ * F_) / 8; idx += 384)
      __builtin_amdgcn_global_load_async_to_lds_b128(
          (ga_v4i*)(s2 + idx), (la_v4i*)(d2 + idx), 0, 0);
#else
    for (int idx = tid; idx < (F_ * KP) / 8; idx += 384) d1[idx] = s1[idx];
    for (int idx = tid; idx < (F_ * F_) / 8; idx += 384) d2[idx] = s2[idx];
#endif
  }
  const float* rbf_row = rbf + (size_t)(b * N_ + i) * (N_ * G_);
  {   // f32 pairs -> packed f16 pairs (50 = 25 float2 per neighbor row)
    const float2* r2 = (const float2*)rbf_row;
    for (int idx = tid; idx < N_ * 25; idx += 384) {
      int j = idx / 25, m = idx - j * 25;
      float2 v = r2[j * 25 + m];
      v2h p; p.x = (half_t)v.x; p.y = (half_t)v.y;
      *(v2h*)&sm.rbfh[j * KP + 2 * m] = p;
    }
    for (int idx = tid; idx < N_ * 7; idx += 384) {   // zero-pad g = 50..63
      int j = idx / 7, m = idx - j * 7;
      *(unsigned int*)&sm.rbfh[j * KP + G_ + 2 * m] = 0u;
    }
  }
  const int* mrow = nmask + (size_t)(b * N_ + i) * N_;
  for (int idx = tid; idx < N_; idx += 384) sm.maskf[idx] = (float)mrow[idx];
  if (tid < F_) {
    sm.brbf1[tid] = b_rbf1[tid];
    sm.brbf2[tid] = b_rbf2[tid];
    sm.agg[tid]   = 0.f;
  }
#if defined(HAS_ASYNC_LDS)
  __builtin_amdgcn_s_wait_asynccnt(0);
#endif
  __syncthreads();

  // ---------------- wave tiling ----------------
  const int wv   = tid >> 5;      // 0..11
  const int lane = tid & 31;
  const int lr   = lane & 15;     // N-index inside a tile (this lane's j col)
  const int hi   = lane >> 4;     // K-half / M-half selector
  const int jp   = wv % 6;        // j-strip pair
  const int fh   = wv / 6;        // f-tile half (4 tiles each)
  const int j0   = (2 * jp)     * 16 + lr;
  const int j1   = (2 * jp + 1) * 16 + lr;

  // GEMM1 (transposed): D = W1^T x rbf^T -> act^T tiles [f x j], K = 64
  // accumulators pre-loaded with b_rbf1 (bias folded into C)
  v8f acc1[2][4];
  #pragma unroll
  for (int ftl = 0; ftl < 4; ++ftl) {
    v8f bias = *(const v8f*)&sm.brbf1[(fh * 4 + ftl) * 16 + hi * 8];
    acc1[0][ftl] = bias;
    acc1[1][ftl] = bias;
  }
  #pragma unroll
  for (int kt = 0; kt < 2; ++kt) {
    const v16h b0 = *(const v16h*)&sm.rbfh[j0 * KP + kt * 32 + hi * 16];
    const v16h b1 = *(const v16h*)&sm.rbfh[j1 * KP + kt * 32 + hi * 16];
    #pragma unroll
    for (int ftl = 0; ftl < 4; ++ftl) {
      const int ft = fh * 4 + ftl;
      const v16h a = *(const v16h*)&sm.w1t[(ft * 16 + lr) * KP + kt * 32 + hi * 16];
      acc1[0][ftl] = __builtin_amdgcn_wmma_f32_16x16x32_f16(
          false, a, false, b0, (short)0, acc1[0][ftl], false, false);
      acc1[1][ftl] = __builtin_amdgcn_wmma_f32_16x16x32_f16(
          false, a, false, b1, (short)0, acc1[1][ftl], false, false);
    }
  }

  // softplus, packed b128 store into row-major acth[j][f]
  #pragma unroll
  for (int s = 0; s < 2; ++s) {
    const int j = (s == 0) ? j0 : j1;
    #pragma unroll
    for (int ftl = 0; ftl < 4; ++ftl) {
      const int ft = fh * 4 + ftl;
      v8h pk;
      #pragma unroll
      for (int v = 0; v < 8; ++v) pk[v] = (half_t)softplus_f(acc1[s][ftl][v]);
      *(v8h*)&sm.acth[j * F_ + ft * 16 + hi * 8] = pk;
    }
  }
  __syncthreads();   // acth rows come from two waves (f-halves)

  // GEMM2 (transposed): D = W2^T x act^T -> msg^T tiles [f x j], K = 128
  v8f acc2[2][4];
  #pragma unroll
  for (int ftl = 0; ftl < 4; ++ftl) {
    v8f bias = *(const v8f*)&sm.brbf2[(fh * 4 + ftl) * 16 + hi * 8];
    acc2[0][ftl] = bias;
    acc2[1][ftl] = bias;
  }
  #pragma unroll
  for (int kt = 0; kt < 4; ++kt) {
    const v16h b0 = *(const v16h*)&sm.acth[j0 * F_ + kt * 32 + hi * 16];
    const v16h b1 = *(const v16h*)&sm.acth[j1 * F_ + kt * 32 + hi * 16];
    #pragma unroll
    for (int ftl = 0; ftl < 4; ++ftl) {
      const int ft = fh * 4 + ftl;
      const v16h a = *(const v16h*)&sm.w2t[(ft * 16 + lr) * F_ + kt * 32 + hi * 16];
      acc2[0][ftl] = __builtin_amdgcn_wmma_f32_16x16x32_f16(
          false, a, false, b0, (short)0, acc2[0][ftl], false, false);
      acc2[1][ftl] = __builtin_amdgcn_wmma_f32_16x16x32_f16(
          false, a, false, b1, (short)0, acc2[1][ftl], false, false);
    }
  }

  // ---- masked nf-product + neighbor reduction (value-splitting butterfly) --
  #pragma unroll
  for (int s = 0; s < 2; ++s) {
    const int   jcol  = (s == 0) ? j0 : j1;
    const float mval  = sm.maskf[jcol];
    const half_t* nfrow = nfh + (size_t)(b * N_ + jcol) * F_;
    #pragma unroll
    for (int ftl = 0; ftl < 4; ++ftl) {
      const int ft = fh * 4 + ftl;
      const v8h nv = *(const v8h*)&nfrow[ft * 16 + hi * 8];
      float p[8];
      #pragma unroll
      for (int v = 0; v < 8; ++v) p[v] = acc2[s][ftl][v] * (float)nv[v] * mval;
      reduce16_accum(p, lane, &sm.agg[ft * 16 + hi * 8]);
    }
  }
  __syncthreads();

  // ---------------- dense_out MLP + residual (row-major f32 matvecs) --------
  if (tid < H_) {
    const float4* wr = (const float4*)wo1t + tid * (F_ / 4);
    const float4* ar = (const float4*)sm.agg;
    float s0 = 0.f, s1 = 0.f, s2 = 0.f, s3 = 0.f;   // 4-way ILP
    #pragma unroll
    for (int k4 = 0; k4 < F_ / 4; ++k4) {
      float4 w = wr[k4]; float4 a = ar[k4];
      s0 = fmaf(a.x, w.x, s0); s1 = fmaf(a.y, w.y, s1);
      s2 = fmaf(a.z, w.z, s2); s3 = fmaf(a.w, w.w, s3);
    }
    sm.h1[tid] = softplus_f(b_o1[tid] + ((s0 + s1) + (s2 + s3)));
  }
  __syncthreads();
  if (tid < H_) {
    const float4* wr = (const float4*)wo2t + tid * (H_ / 4);
    const float4* hr = (const float4*)sm.h1;
    float s0 = 0.f, s1 = 0.f, s2 = 0.f, s3 = 0.f;
    #pragma unroll
    for (int k4 = 0; k4 < H_ / 4; ++k4) {
      float4 w = wr[k4]; float4 h = hr[k4];
      s0 = fmaf(h.x, w.x, s0); s1 = fmaf(h.y, w.y, s1);
      s2 = fmaf(h.z, w.z, s2); s3 = fmaf(h.w, w.w, s3);
    }
    const size_t o = (size_t)(b * N_ + i) * H_ + tid;
    out[o] = features[o] + b_o2[tid] + ((s0 + s1) + (s2 + s3));
  }
}

// --------------------------------- launch ----------------------------------
extern "C" void kernel_launch(void* const* d_in, const int* in_sizes, int n_in,
                              void* d_out, int out_size, void* d_ws, size_t ws_size,
                              hipStream_t stream) {
  const float* features = (const float*)d_in[0];
  const float* rbf      = (const float*)d_in[1];
  const int*   nmask    = (const int*)  d_in[2];
  const float* W_rbf1   = (const float*)d_in[3];
  const float* b_rbf1   = (const float*)d_in[4];
  const float* W_rbf2   = (const float*)d_in[5];
  const float* b_rbf2   = (const float*)d_in[6];
  const float* W_f      = (const float*)d_in[7];
  const float* b_f      = (const float*)d_in[8];
  const float* W_o1     = (const float*)d_in[9];
  const float* b_o1     = (const float*)d_in[10];
  const float* W_o2     = (const float*)d_in[11];
  const float* b_o2     = (const float*)d_in[12];
  float* out = (float*)d_out;

  // workspace layout (~944 KB): f16 nf + f16 transposed filter weights +
  // f32 transposed output-MLP weights
  half_t* nfh  = (half_t*)d_ws;
  half_t* w1t  = nfh + (size_t)B_ * N_ * F_;
  half_t* w2t  = w1t + (size_t)F_ * KP;
  float*  wo1t = (float*)(w2t + (size_t)F_ * F_);
  float*  wo2t = wo1t + (size_t)H_ * F_;

  prep_weights<<<1, 256, 0, stream>>>(W_rbf1, W_rbf2, W_o1, W_o2,
                                      w1t, w2t, wo1t, wo2t);
  prep_nf<<<B_ * N_, H_, 0, stream>>>(features, W_f, b_f, nfh);
  interaction_main<<<dim3(N_, B_), 384, 0, stream>>>(
      features, rbf, nmask, b_rbf1, b_rbf2,
      b_o1, b_o2, nfh, w1t, w2t, wo1t, wo2t, out);
}
